// ScaledDotProductAttention_43250320670790
// MI455X (gfx1250) — matrix operand order repaired and checked by reading, not compiled, not run on hardware
//
#include <hip/hip_runtime.h>
#include <hip/hip_bf16.h>

#define B_SZ 4
#define NQ   4096
#define NVK  4096
#define DM   256
#define DK   64
#define DV   64

typedef __attribute__((ext_vector_type(16))) __bf16 v16bf;
typedef __attribute__((ext_vector_type(8)))  __bf16 v8bf;
typedef __attribute__((ext_vector_type(8)))  float  v8f;

#define WMMA_BF16(Am, Bm, Cm) \
  __builtin_amdgcn_wmma_f32_16x16x32_bf16(false, (Am), false, (Bm), (short)0, (Cm), false, false)

// ---- WMMA fragment loaders (bf16, row-major source) -------------------------
// A operand, 16x32 (MxK): lane r=lane&15 holds row M=r; hi=lane>>4 selects
// K in {hi*8..hi*8+7} U {16+hi*8..16+hi*8+7}  (ISA 7.12.2, 16-bit A 16x32)
__device__ __forceinline__ v16bf load_a_frag(const __bf16* __restrict__ base, int ld, int lane) {
  const int r = lane & 15, hi = lane >> 4;
  const __bf16* p = base + (size_t)r * ld + hi * 8;
  v8bf a = *(const v8bf*)(p);
  v8bf b = *(const v8bf*)(p + 16);
  v16bf f;
#pragma unroll
  for (int j = 0; j < 8; ++j) { f[j] = a[j]; f[j + 8] = b[j]; }
  return f;
}
// B operand, 32x16 (KxN): lane r holds column N=r; hi selects K in {hi*8..hi*8+7} U {16+hi*8..16+hi*8+7}
__device__ __forceinline__ v16bf load_b_frag(const __bf16* __restrict__ base, int ld, int lane) {
  const int r = lane & 15, hi = lane >> 4;
  const __bf16* p = base + (size_t)r * ld + hi * 8;
  v8bf a = *(const v8bf*)(p);
  v8bf b = *(const v8bf*)(p + 16);
  v16bf f;
#pragma unroll
  for (int j = 0; j < 8; ++j) { f[j] = a[j]; f[j + 8] = b[j]; }
  return f;
}

// ---- tiny prep: W [256,64] f32 -> Wt [64,256] bf16 --------------------------
__global__ void transpose_w_kernel(const float* __restrict__ W, __bf16* __restrict__ Wt) {
  int i = blockIdx.x * blockDim.x + threadIdx.x;
  if (i < DM * DK) {
    int k = i >> 6, c = i & 63;
    Wt[(size_t)c * DM + k] = (__bf16)W[i];
  }
}

// ---- projection: out[r, :64] = bf16( X[r, :256] @ W + bias ), via WMMA ------
__global__ __launch_bounds__(256) void proj_wmma_kernel(
    const float* __restrict__ X, const __bf16* __restrict__ Wt,
    const float* __restrict__ bias, __bf16* __restrict__ out, int nrows) {
  const int lane = threadIdx.x & 31;
  const int wave = threadIdx.x >> 5;
  const int row0 = (blockIdx.x * 8 + wave) * 16;
  if (row0 >= nrows) return;
  const int r = lane & 15, hi = lane >> 4;
  const float* xrow = X + (size_t)(row0 + r) * DM + hi * 8;
  v8f acc[4] = {};

  // prologue: A f32 slice for kk = 0
  float4 a0 = *(const float4*)(xrow);
  float4 a1 = *(const float4*)(xrow + 4);
  float4 b0 = *(const float4*)(xrow + 16);
  float4 b1 = *(const float4*)(xrow + 20);

#pragma unroll 1
  for (int kk = 0; kk < DM; kk += 32) {
    // batch the 4 weight B-fragments up front
    v16bf Bf0 = load_b_frag(Wt + (size_t)0  * DM + kk, DM, lane);
    v16bf Bf1 = load_b_frag(Wt + (size_t)16 * DM + kk, DM, lane);
    v16bf Bf2 = load_b_frag(Wt + (size_t)32 * DM + kk, DM, lane);
    v16bf Bf3 = load_b_frag(Wt + (size_t)48 * DM + kk, DM, lane);
    // prefetch next A f32 slice (wrap on last iter; harmless)
    const float* nx = xrow + ((kk + 32) & (DM - 1));
    float4 na0 = *(const float4*)(nx);
    float4 na1 = *(const float4*)(nx + 4);
    float4 nb0 = *(const float4*)(nx + 16);
    float4 nb1 = *(const float4*)(nx + 20);
    // convert current A to bf16 fragment
    v16bf A;
    A[0]=(__bf16)a0.x; A[1]=(__bf16)a0.y; A[2]=(__bf16)a0.z; A[3]=(__bf16)a0.w;
    A[4]=(__bf16)a1.x; A[5]=(__bf16)a1.y; A[6]=(__bf16)a1.z; A[7]=(__bf16)a1.w;
    A[8]=(__bf16)b0.x; A[9]=(__bf16)b0.y; A[10]=(__bf16)b0.z; A[11]=(__bf16)b0.w;
    A[12]=(__bf16)b1.x; A[13]=(__bf16)b1.y; A[14]=(__bf16)b1.z; A[15]=(__bf16)b1.w;
    acc[0] = WMMA_BF16(A, Bf0, acc[0]);
    acc[1] = WMMA_BF16(A, Bf1, acc[1]);
    acc[2] = WMMA_BF16(A, Bf2, acc[2]);
    acc[3] = WMMA_BF16(A, Bf3, acc[3]);
    a0 = na0; a1 = na1; b0 = nb0; b1 = nb1;
  }
#pragma unroll
  for (int t = 0; t < 4; ++t) {
    float bv = bias[t * 16 + r];
#pragma unroll
    for (int rr = 0; rr < 8; ++rr) {
      int row = row0 + rr + 8 * hi;                 // C/D layout: M = rr + 8*hi
      out[(size_t)row * DK + t * 16 + r] = (__bf16)(acc[t][rr] + bv);
    }
  }
}

// ---- vb [B,NVK,64] -> vtb [B,64,NVK] (bf16, LDS-tiled transpose) ------------
__global__ __launch_bounds__(256) void transpose_v_kernel(
    const __bf16* __restrict__ vb, __bf16* __restrict__ vtb) {
  __shared__ __bf16 tile[64 * 65];
  const int b  = blockIdx.x / (NVK / 64);
  const int n0 = (blockIdx.x % (NVK / 64)) * 64;
  const int t  = threadIdx.x;
#pragma unroll
  for (int j = 0; j < 16; ++j) {
    int idx = j * 256 + t, row = idx >> 6, col = idx & 63;
    tile[row * 65 + col] = vb[((size_t)(b * NVK + n0 + row)) * DK + col];
  }
  __syncthreads();
#pragma unroll
  for (int j = 0; j < 16; ++j) {
    int idx = j * 256 + t, row = idx >> 6, col = idx & 63; // row = v-dim, col = key
    vtb[((size_t)(b * DV + row)) * NVK + n0 + col] = tile[col * 65 + row];
  }
}

// ---- fused attention: S^T = K.Q^T, online softmax, attn write, ctx^T = V^T.P^T
__global__ __launch_bounds__(256) void attn_kernel(
    const __bf16* __restrict__ qb, const __bf16* __restrict__ kb,
    const __bf16* __restrict__ vtb, float* __restrict__ ctx_out,
    float* __restrict__ attn_out) {
  __shared__ float lds[8 * 16 * 68];
  const int lane = threadIdx.x & 31;
  const int wave = threadIdx.x >> 5;
  const int tile = blockIdx.x * 8 + wave;            // 128 blocks * 8 waves = 1024 tiles
  const int b  = tile / (NQ / 16);
  const int q0 = (tile % (NQ / 16)) * 16;
  const int n  = lane & 15;                          // query column within tile
  const int hi = lane >> 4;
  float* L = lds + wave * (16 * 68);
  const float sc = 0.125f;                           // 1/sqrt(64)

  // Q^T as B operand (loop invariant): dk 0..31 and 32..63
  const __bf16* qbase = qb + (size_t)(b * NQ + q0) * DK;
  const v16bf qB0 = load_b_frag(qbase, DK, lane);
  const v16bf qB1 = load_b_frag(qbase + 32, DK, lane);
  const __bf16* kbB = kb + (size_t)b * NVK * DK;
  const __bf16* vtB = vtb + (size_t)b * DV * NVK;

  // -------- pass 1: online row max / sum, 32 keys/iter, pipelined -----------
  float m = -3.0e38f, l = 0.f;
  {
    v16bf c00 = load_a_frag(kbB, DK, lane);
    v16bf c01 = load_a_frag(kbB + 32, DK, lane);
    v16bf c10 = load_a_frag(kbB + 16 * DK, DK, lane);
    v16bf c11 = load_a_frag(kbB + 16 * DK + 32, DK, lane);
#pragma unroll 1
    for (int k0 = 0; k0 < NVK; k0 += 32) {
      // prefetch next key block (wrap on last iter)
      const __bf16* np = kbB + (size_t)((k0 + 32) & (NVK - 1)) * DK;
      v16bf n00 = load_a_frag(np, DK, lane);
      v16bf n01 = load_a_frag(np + 32, DK, lane);
      v16bf n10 = load_a_frag(np + 16 * DK, DK, lane);
      v16bf n11 = load_a_frag(np + 16 * DK + 32, DK, lane);
      // two independent accumulator tiles (interleaved WMMAs hide hazards)
      v8f a0 = {}, a1 = {};
      a0 = WMMA_BF16(c00, qB0, a0);
      a1 = WMMA_BF16(c10, qB0, a1);
      a0 = WMMA_BF16(c01, qB1, a0);
      a1 = WMMA_BF16(c11, qB1, a1);
      float tmax = a0[0] * sc;
#pragma unroll
      for (int rr = 1; rr < 8; ++rr) tmax = fmaxf(tmax, a0[rr] * sc);
#pragma unroll
      for (int rr = 0; rr < 8; ++rr) tmax = fmaxf(tmax, a1[rr] * sc);
      tmax = fmaxf(tmax, __shfl_xor(tmax, 16, 32));
      float mnew = fmaxf(m, tmax);
      float s16 = 0.f;
#pragma unroll
      for (int rr = 0; rr < 8; ++rr)
        s16 += __expf(a0[rr] * sc - mnew) + __expf(a1[rr] * sc - mnew);
      s16 += __shfl_xor(s16, 16, 32);
      l = l * __expf(m - mnew) + s16;
      m = mnew;
      c00 = n00; c01 = n01; c10 = n10; c11 = n11;
    }
  }
  const float inv_l = 1.0f / l;

  // -------- pass 2: normalize, write attn (coalesced via LDS), P.V ----------
  v8f ctx0 = {}, ctx1 = {}, ctx2 = {}, ctx3 = {};
  {
    v16bf c00 = load_a_frag(kbB, DK, lane);
    v16bf c01 = load_a_frag(kbB + 32, DK, lane);
    v16bf c10 = load_a_frag(kbB + 16 * DK, DK, lane);
    v16bf c11 = load_a_frag(kbB + 16 * DK + 32, DK, lane);
#pragma unroll 1
    for (int k0 = 0; k0 < NVK; k0 += 32) {
      // V^T fragments issued first: consumed last (after softmax/store section)
      v16bf vA0 = load_a_frag(vtB + (size_t)0  * NVK + k0, NVK, lane);
      v16bf vA1 = load_a_frag(vtB + (size_t)16 * NVK + k0, NVK, lane);
      v16bf vA2 = load_a_frag(vtB + (size_t)32 * NVK + k0, NVK, lane);
      v16bf vA3 = load_a_frag(vtB + (size_t)48 * NVK + k0, NVK, lane);
      // scores for this 32-key block
      v8f a0 = {}, a1 = {};
      a0 = WMMA_BF16(c00, qB0, a0);
      a1 = WMMA_BF16(c10, qB0, a1);
      a0 = WMMA_BF16(c01, qB1, a0);
      a1 = WMMA_BF16(c11, qB1, a1);
      // prefetch next key block (overlaps exp/store tail)
      const __bf16* np = kbB + (size_t)((k0 + 32) & (NVK - 1)) * DK;
      v16bf n00 = load_a_frag(np, DK, lane);
      v16bf n01 = load_a_frag(np + 32, DK, lane);
      v16bf n10 = load_a_frag(np + 16 * DK, DK, lane);
      v16bf n11 = load_a_frag(np + 16 * DK + 32, DK, lane);
      // normalize + stage P (query-major) in LDS
      float pv0[8], pv1[8];
#pragma unroll
      for (int rr = 0; rr < 8; ++rr) {
        float p0 = __expf(a0[rr] * sc - m) * inv_l;
        float p1 = __expf(a1[rr] * sc - m) * inv_l;
        pv0[rr] = p0; pv1[rr] = p1;
        L[n * 68 + hi * 8 + rr]      = p0;
        L[n * 68 + 16 + hi * 8 + rr] = p1;
      }
      asm volatile("s_wait_dscnt 0" ::: "memory");
      // coalesced attn store: 16 rows x 32 floats = 128 chunks of 16B
      float* aout = attn_out + (size_t)(b * NQ + q0) * NVK + k0;
#pragma unroll
      for (int j = 0; j < 4; ++j) {
        int idx = j * 32 + lane, row = idx >> 3, c = idx & 7;
        float4 v = *(const float4*)&L[row * 68 + c * 4];
        *(float4*)&aout[(size_t)row * NVK + c * 4] = v;
      }
      // pack P^T into bf16 B operand: lane n needs keys {0..7,16..23}, lane n+16 {8..15,24..31}
      v16bf pB;
#pragma unroll
      for (int rr = 0; rr < 8; ++rr) {
        float oth0 = __shfl_xor(pv0[rr], 16, 32);
        float oth1 = __shfl_xor(pv1[rr], 16, 32);
        pB[rr]     = (__bf16)(pv0[rr]);
        pB[8 + rr] = (__bf16)(pv1[rr]);
      }
      // ctx^T accumulation (vA loads long since complete)
      ctx0 = WMMA_BF16(vA0, pB, ctx0);
      ctx1 = WMMA_BF16(vA1, pB, ctx1);
      ctx2 = WMMA_BF16(vA2, pB, ctx2);
      ctx3 = WMMA_BF16(vA3, pB, ctx3);
      c00 = n00; c01 = n01; c10 = n10; c11 = n11;
    }
  }

  // -------- store context via LDS transpose (coalesced b128) ----------------
#pragma unroll
  for (int rr = 0; rr < 8; ++rr) {
    L[n * 68 +  0 + hi * 8 + rr] = ctx0[rr];
    L[n * 68 + 16 + hi * 8 + rr] = ctx1[rr];
    L[n * 68 + 32 + hi * 8 + rr] = ctx2[rr];
    L[n * 68 + 48 + hi * 8 + rr] = ctx3[rr];
  }
  asm volatile("s_wait_dscnt 0" ::: "memory");
  float* cout = ctx_out + (size_t)(b * NQ + q0) * DV;
#pragma unroll
  for (int j = 0; j < 8; ++j) {
    int idx = j * 32 + lane, row = idx >> 4, c = idx & 15;
    float4 v = *(const float4*)&L[row * 68 + c * 4];
    *(float4*)&cout[(size_t)row * DV + c * 4] = v;
  }
}

extern "C" void kernel_launch(void* const* d_in, const int* in_sizes, int n_in,
                              void* d_out, int out_size, void* d_ws, size_t ws_size,
                              hipStream_t stream) {
  (void)in_sizes; (void)n_in; (void)out_size;
  if (ws_size < (9u << 20)) return;   // need ~8.2 MB scratch

  const float* Q  = (const float*)d_in[0];
  const float* K  = (const float*)d_in[1];
  const float* V  = (const float*)d_in[2];
  const float* Wq = (const float*)d_in[3];
  const float* bq = (const float*)d_in[4];
  const float* Wk = (const float*)d_in[5];
  const float* bk = (const float*)d_in[6];
  const float* Wv = (const float*)d_in[7];
  const float* bv = (const float*)d_in[8];

  char* ws = (char*)d_ws;
  __bf16* qb  = (__bf16*)(ws);                               // [B,NQ,64]  bf16, 2 MB
  __bf16* kb  = (__bf16*)(ws + (2u << 20));                  // [B,NVK,64] bf16, 2 MB
  __bf16* vb  = (__bf16*)(ws + (4u << 20));                  // [B,NVK,64] bf16, 2 MB
  __bf16* vtb = (__bf16*)(ws + (6u << 20));                  // [B,64,NVK] bf16, 2 MB
  __bf16* wtq = (__bf16*)(ws + (8u << 20));                  // [64,256] bf16
  __bf16* wtk = (__bf16*)(ws + (8u << 20) + (64u << 10));
  __bf16* wtv = (__bf16*)(ws + (8u << 20) + (128u << 10));

  transpose_w_kernel<<<64, 256, 0, stream>>>(Wq, wtq);
  transpose_w_kernel<<<64, 256, 0, stream>>>(Wk, wtk);
  transpose_w_kernel<<<64, 256, 0, stream>>>(Wv, wtv);

  const int nrows = B_SZ * NQ;                               // 16384 rows each
  proj_wmma_kernel<<<128, 256, 0, stream>>>(Q, wtq, bq, qb, nrows);
  proj_wmma_kernel<<<128, 256, 0, stream>>>(K, wtk, bk, kb, nrows);
  proj_wmma_kernel<<<128, 256, 0, stream>>>(V, wtv, bv, vb, nrows);

  transpose_v_kernel<<<B_SZ * (NVK / 64), 256, 0, stream>>>(vb, vtb);

  float* ctx_out  = (float*)d_out;                           // [B,NQ,64]
  float* attn_out = ctx_out + (size_t)B_SZ * NQ * DV;        // [B,NQ,NVK]
  attn_kernel<<<128, 256, 0, stream>>>(qb, kb, vtb, ctx_out, attn_out);
}
